// ScaledDotProductAttention_75015898792335
// MI455X (gfx1250) — matrix operand order repaired and checked by reading, not compiled, not run on hardware
//
#include <hip/hip_runtime.h>

#define S_LEN 2048
#define D_K   64
#define N_H   16
#define N_B   2
#define SCALE 0.125f   // 1/sqrt(64)

typedef _Float16 v16h __attribute__((ext_vector_type(16)));
typedef _Float16 v8h  __attribute__((ext_vector_type(8)));
typedef _Float16 v4h  __attribute__((ext_vector_type(4)));
typedef float    v8f  __attribute__((ext_vector_type(8)));

// LDS pitches (halfs): multiples of 8 halfs (16B) for aligned ds_load_b128,
// chosen so strided row gathers spread across all 64 banks.
#define QP 72
#define KP 72
#define VP 40
#define PP 40

// A-matrix 16x32 f16 fragment: lane half selects K-subgroup {0-7,16-23} / {8-15,24-31}
__device__ __forceinline__ v16h fragA(const _Float16* rb, int lane) {
  const int off = (lane & 16) ? 8 : 0;
  v8h lo = *(const v8h*)(rb + off);
  v8h hi = *(const v8h*)(rb + off + 16);
  return __builtin_shufflevector(lo, hi, 0,1,2,3,4,5,6,7,8,9,10,11,12,13,14,15);
}
// B-matrix 32x16 f16 fragment: lane half selects K-subgroup {0-7,16-23} / {8-15,24-31}
__device__ __forceinline__ v16h fragB(const _Float16* rb, int lane) {
  const int off = (lane & 16) ? 8 : 0;
  v8h lo = *(const v8h*)(rb + off);
  v8h hi = *(const v8h*)(rb + off + 16);
  return __builtin_shufflevector(lo, hi, 0,1,2,3,4,5,6,7,8,9,10,11,12,13,14,15);
}

__global__ __launch_bounds__(128)
void attn_fused_kernel(const float* __restrict__ Q, const float* __restrict__ K,
                       const float* __restrict__ V, const float* __restrict__ Msk,
                       const float* __restrict__ Dst, float* __restrict__ Out)
{
  __shared__ __align__(16) _Float16 sQ[4 * 16 * QP];
  __shared__ __align__(16) _Float16 sK[32 * KP];
  __shared__ __align__(16) _Float16 sVt[64 * VP];   // transposed: [dk][kvrow]
  __shared__ __align__(16) _Float16 sP[4 * 16 * PP];

  const int tid  = threadIdx.x;
  const int w    = tid >> 5;
  const int lane = tid & 31;
  const int nl   = lane & 15;
  const int hb8  = (lane & 16) ? 8 : 0;

  const int bh = blockIdx.x >> 5;   // b*H + h  (0..31)
  const int qb = blockIdx.x & 31;   // q block of 64 rows
  const int b  = bh >> 4;
  // wave-uniform first q row: force into SGPR so all global bases scalarize
  const int q0 = __builtin_amdgcn_readfirstlane(qb * 64 + w * 16);

  const float* qp  = Q + ((size_t)bh * S_LEN + q0) * D_K;
  const float* kbp = K + (size_t)bh * S_LEN * D_K;
  const float* vbp = V + (size_t)bh * S_LEN * D_K;
  const float* mQ  = Msk + (size_t)b * S_LEN * S_LEN + (size_t)q0 * S_LEN;
  const float* dQ  = Dst + (size_t)b * S_LEN * S_LEN + (size_t)q0 * S_LEN;
  float* outQ  = Out + ((size_t)bh * S_LEN + q0) * D_K;
  float* attnQ = Out + (size_t)N_B * N_H * S_LEN * D_K
                     + (size_t)bh * S_LEN * S_LEN + (size_t)q0 * S_LEN;

  _Float16* sQw = sQ + w * 16 * QP;
  _Float16* sPw = sP + w * 16 * PP;

  // stage this wave's Q tile, f32 -> f16
  for (int i = lane; i < 16 * D_K; i += 32)
    sQw[(i >> 6) * QP + (i & 63)] = (_Float16)qp[i];
  __syncthreads();

  const v16h qa0 = fragA(sQw + nl * QP +  0, lane);   // K dims 0..31
  const v16h qa1 = fragA(sQw + nl * QP + 32, lane);   // K dims 32..63

  const float4* kp4 = (const float4*)kbp;   // 16 float4 per kv row
  const float4* vp4 = (const float4*)vbp;

  float lsum[8];
  #pragma unroll
  for (int r = 0; r < 8; ++r) lsum[r] = 0.f;

  // ---------------- pass 1: per-row sum of exp(scores) ----------------
  float4 kreg[4];
  #pragma unroll
  for (int e = 0; e < 4; ++e) kreg[e] = kp4[e * 128 + tid];   // tile jj=0

  #pragma unroll 1
  for (int jj = 0; jj < S_LEN; jj += 32) {
    // store pipelined K tile (f32 regs -> f16 LDS), 8B ds stores
    #pragma unroll
    for (int e = 0; e < 4; ++e) {
      const int idx4 = e * 128 + tid;
      const int row = idx4 >> 4, col = (idx4 & 15) * 4;
      float4 f = kreg[e];
      *(v4h*)(sK + row * KP + col) =
          (v4h){(_Float16)f.x, (_Float16)f.y, (_Float16)f.z, (_Float16)f.w};
    }
    __syncthreads();

    // prefetch next K tile into regs (overlaps with everything below)
    if (jj + 32 < S_LEN) {
      #pragma unroll
      for (int e = 0; e < 4; ++e) kreg[e] = kp4[(jj + 32) * 16 + e * 128 + tid];
    }

    // batch mask/dist loads for this block (issued before ds/wmma work)
    float mv0[8], mv1[8], dv0[8], dv1[8];
    #pragma unroll
    for (int r = 0; r < 8; ++r) {
      const int ro = (r + hb8) * S_LEN + jj + nl;
      mv0[r] = mQ[ro]; mv1[r] = mQ[ro + 16];
      dv0[r] = dQ[ro]; dv1[r] = dQ[ro + 16];
    }

    v16h b00 = fragB(sK + nl * KP +  0, lane);
    v16h b01 = fragB(sK + nl * KP + 32, lane);
    v16h b10 = fragB(sK + (16 + nl) * KP +  0, lane);
    v16h b11 = fragB(sK + (16 + nl) * KP + 32, lane);

    v8f c0 = {0.f,0.f,0.f,0.f,0.f,0.f,0.f,0.f};
    v8f c1 = {0.f,0.f,0.f,0.f,0.f,0.f,0.f,0.f};
    c0 = __builtin_amdgcn_wmma_f32_16x16x32_f16(false, qa0, false, b00, (short)0, c0, false, false);
    c0 = __builtin_amdgcn_wmma_f32_16x16x32_f16(false, qa1, false, b01, (short)0, c0, false, false);
    c1 = __builtin_amdgcn_wmma_f32_16x16x32_f16(false, qa0, false, b10, (short)0, c1, false, false);
    c1 = __builtin_amdgcn_wmma_f32_16x16x32_f16(false, qa1, false, b11, (short)0, c1, false, false);

    #pragma unroll
    for (int r = 0; r < 8; ++r) {
      float s0 = fminf(dv0[r] * fmaf(mv0[r], -1e9f, c0[r] * SCALE), 30.f);
      float s1 = fminf(dv1[r] * fmaf(mv1[r], -1e9f, c1[r] * SCALE), 30.f);
      lsum[r] += __expf(s0) + __expf(s1);
    }
    __syncthreads();
  }

  // reduce partial sums across the 16 lanes sharing each row
  float invl[8];
  #pragma unroll
  for (int r = 0; r < 8; ++r) {
    float v = lsum[r];
    v += __shfl_xor(v, 1, 32);
    v += __shfl_xor(v, 2, 32);
    v += __shfl_xor(v, 4, 32);
    v += __shfl_xor(v, 8, 32);
    invl[r] = 1.0f / v;
  }

  v8f o0 = {0.f,0.f,0.f,0.f,0.f,0.f,0.f,0.f};
  v8f o1 = o0, o2 = o0, o3 = o0;

  // ---------------- pass 2: recompute scores, write attn, P @ V ----------------
  float4 vreg[4];
  #pragma unroll
  for (int e = 0; e < 4; ++e) kreg[e] = kp4[e * 128 + tid];
  #pragma unroll
  for (int e = 0; e < 4; ++e) vreg[e] = vp4[e * 128 + tid];

  #pragma unroll 1
  for (int jj = 0; jj < S_LEN; jj += 32) {
    // store pipelined K/V tiles to LDS (V transposed)
    #pragma unroll
    for (int e = 0; e < 4; ++e) {
      const int idx4 = e * 128 + tid;
      const int row = idx4 >> 4, col = (idx4 & 15) * 4;
      float4 f = kreg[e];
      *(v4h*)(sK + row * KP + col) =
          (v4h){(_Float16)f.x, (_Float16)f.y, (_Float16)f.z, (_Float16)f.w};
      float4 g = vreg[e];
      sVt[(col + 0) * VP + row] = (_Float16)g.x;
      sVt[(col + 1) * VP + row] = (_Float16)g.y;
      sVt[(col + 2) * VP + row] = (_Float16)g.z;
      sVt[(col + 3) * VP + row] = (_Float16)g.w;
    }
    __syncthreads();

    if (jj + 32 < S_LEN) {
      #pragma unroll
      for (int e = 0; e < 4; ++e) kreg[e] = kp4[(jj + 32) * 16 + e * 128 + tid];
      #pragma unroll
      for (int e = 0; e < 4; ++e) vreg[e] = vp4[(jj + 32) * 16 + e * 128 + tid];
    }

    float mv0[8], mv1[8], dv0[8], dv1[8];
    #pragma unroll
    for (int r = 0; r < 8; ++r) {
      const int ro = (r + hb8) * S_LEN + jj + nl;
      mv0[r] = mQ[ro]; mv1[r] = mQ[ro + 16];
      dv0[r] = dQ[ro]; dv1[r] = dQ[ro + 16];
    }

    v16h b00 = fragB(sK + nl * KP +  0, lane);
    v16h b01 = fragB(sK + nl * KP + 32, lane);
    v16h b10 = fragB(sK + (16 + nl) * KP +  0, lane);
    v16h b11 = fragB(sK + (16 + nl) * KP + 32, lane);

    v8f c0 = {0.f,0.f,0.f,0.f,0.f,0.f,0.f,0.f};
    v8f c1 = {0.f,0.f,0.f,0.f,0.f,0.f,0.f,0.f};
    c0 = __builtin_amdgcn_wmma_f32_16x16x32_f16(false, qa0, false, b00, (short)0, c0, false, false);
    c0 = __builtin_amdgcn_wmma_f32_16x16x32_f16(false, qa1, false, b01, (short)0, c0, false, false);
    c1 = __builtin_amdgcn_wmma_f32_16x16x32_f16(false, qa0, false, b10, (short)0, c1, false, false);
    c1 = __builtin_amdgcn_wmma_f32_16x16x32_f16(false, qa1, false, b11, (short)0, c1, false, false);

    #pragma unroll
    for (int r = 0; r < 8; ++r) {
      const int mrow = r + hb8;
      const int ro = mrow * S_LEN + jj + nl;
      float s0 = fminf(dv0[r] * fmaf(mv0[r], -1e9f, c0[r] * SCALE), 30.f);
      float s1 = fminf(dv1[r] * fmaf(mv1[r], -1e9f, c1[r] * SCALE), 30.f);
      float p0 = __expf(s0) * invl[r];
      float p1 = __expf(s1) * invl[r];
      __builtin_nontemporal_store(p0, &attnQ[ro]);        // streaming 512MB output:
      __builtin_nontemporal_store(p1, &attnQ[ro + 16]);   // keep mask/Dist in L2
      sPw[mrow * PP + nl]      = (_Float16)p0;
      sPw[mrow * PP + nl + 16] = (_Float16)p1;
    }
    __syncthreads();

    v16h ap = fragA(sPw + nl * PP, lane);                 // P tile 16x32 (K = kv rows)
    v16h v0 = fragB(sVt + ( 0 + nl) * VP, lane);
    v16h v1 = fragB(sVt + (16 + nl) * VP, lane);
    v16h v2 = fragB(sVt + (32 + nl) * VP, lane);
    v16h v3 = fragB(sVt + (48 + nl) * VP, lane);
    o0 = __builtin_amdgcn_wmma_f32_16x16x32_f16(false, ap, false, v0, (short)0, o0, false, false);
    o1 = __builtin_amdgcn_wmma_f32_16x16x32_f16(false, ap, false, v1, (short)0, o1, false, false);
    o2 = __builtin_amdgcn_wmma_f32_16x16x32_f16(false, ap, false, v2, (short)0, o2, false, false);
    o3 = __builtin_amdgcn_wmma_f32_16x16x32_f16(false, ap, false, v3, (short)0, o3, false, false);
    __syncthreads();
  }

  #pragma unroll
  for (int r = 0; r < 8; ++r) {
    const int mrow = r + hb8;
    __builtin_nontemporal_store(o0[r], &outQ[mrow * D_K +  0 + nl]);
    __builtin_nontemporal_store(o1[r], &outQ[mrow * D_K + 16 + nl]);
    __builtin_nontemporal_store(o2[r], &outQ[mrow * D_K + 32 + nl]);
    __builtin_nontemporal_store(o3[r], &outQ[mrow * D_K + 48 + nl]);
  }
}

extern "C" void kernel_launch(void* const* d_in, const int* in_sizes, int n_in,
                              void* d_out, int out_size, void* d_ws, size_t ws_size,
                              hipStream_t stream) {
  (void)in_sizes; (void)n_in; (void)out_size; (void)d_ws; (void)ws_size;
  const float* Q = (const float*)d_in[0];
  const float* K = (const float*)d_in[1];
  const float* V = (const float*)d_in[2];
  const float* M = (const float*)d_in[3];
  const float* D = (const float*)d_in[4];
  float* Out = (float*)d_out;
  // grid: (B*H) * (S/64) workgroups, 4 waves each (one 16-row q tile per wave)
  dim3 grid(N_B * N_H * (S_LEN / 64));
  attn_fused_kernel<<<grid, 128, 0, stream>>>(Q, K, V, M, D, Out);
}
